// Model_15625091023401
// MI455X (gfx1250) — matrix operand order repaired and checked by reading, not compiled, not run on hardware
//
#include <hip/hip_runtime.h>
#include <cstdint>
#include <cstddef>

// ---------------------------------------------------------------------------
// Types for CDNA5 WMMA (gfx1250, wave32)
// ---------------------------------------------------------------------------
typedef __attribute__((ext_vector_type(16))) _Float16     v16h;
typedef __attribute__((ext_vector_type(8)))  float        v8f;
typedef __attribute__((ext_vector_type(4)))  unsigned int u32x4;

// ---------------------------------------------------------------------------
// Gather-GEMM-scatter conv for ONE rulebook offset.
//   Out[oi[r]][colOff + n] += sum_k X[ii[r]][k] * Wt[n][k]
// X  : f16, row-major, ld = ldX (Cin padded to multiple of 32, zero-padded)
// Wt : f16, TRANSPOSED weights, row n = output channel, ld = ldX, zero-padded
// oi is unique within one offset -> plain read-add-store (no atomics).
// One wave computes a 16(pairs) x (16*NT)(cout) tile: the gathered A fragment
// is loaded once and reused for NT WMMAs (amortizes the random-gather VMEM).
// A fragment (16x32, MxK): lane r (<16) holds row r, K = kk+{0..7,16..23};
// lane r+16 holds row r, K = kk+{8..15,24..31} -> two b128 loads per lane.
// B fragment (32x16, KxN): lane n (<16) holds col n, K = kk+{0..7,16..23};
// lane n+16 holds col n, K = kk+{8..15,24..31} -> two b128 loads per lane.
// ---------------------------------------------------------------------------
template <int NT>
__global__ __launch_bounds__(32)
void gsconv_wmma_kernel(const _Float16* __restrict__ X, int ldX,
                        const _Float16* __restrict__ Wt,
                        const int* __restrict__ ii, const int* __restrict__ oi,
                        int npairs,
                        float* __restrict__ Out, int ldOut, int colOff)
{
  const int lane = threadIdx.x;
  const int tm   = blockIdx.x;               // pair tile
  const int tn0  = blockIdx.y * NT;          // first cout tile
  const int half = (lane >= 16) ? 1 : 0;
  const int l15  = lane & 15;

  int  row    = tm * 16 + l15;
  bool rvalid = row < npairs;
  int  g      = ii[rvalid ? row : (npairs - 1)];   // clamped; stores are masked

  const _Float16* xrow = X  + (long long)g * ldX;
  const _Float16* wcol = Wt + (long long)(tn0 * 16 + l15) * ldX;
  const long long wstride = (long long)16 * ldX;   // one cout tile of Wt rows

  v8f zerov = {};
  v8f acc[NT];
#pragma unroll
  for (int t = 0; t < NT; ++t) acc[t] = zerov;

  for (int kk = 0; kk < ldX; kk += 32) {
    union { v16h v; u32x4 q[2]; } a;
    int ka = kk + (half ? 8 : 0);
    a.q[0] = *(const u32x4*)(xrow + ka);        // K = ka .. ka+7
    a.q[1] = *(const u32x4*)(xrow + ka + 16);   // K = ka+16 .. ka+23
    int kb = kk + (half ? 8 : 0);
#pragma unroll
    for (int t = 0; t < NT; ++t) {
      union { v16h v; u32x4 q[2]; } b;
      const _Float16* wp = wcol + (long long)t * wstride + kb;
      b.q[0] = *(const u32x4*)(wp);             // K = kb .. kb+7
      b.q[1] = *(const u32x4*)(wp + 16);        // K = kb+16 .. kb+23
      acc[t] = __builtin_amdgcn_wmma_f32_16x16x32_f16(
          /*neg_a=*/false, a.v, /*neg_b=*/false, b.v,
          /*c_mod=*/(short)0, acc[t], /*reuse_a=*/false, /*reuse_b=*/false);
    }
  }

  // C/D layout: VGPR v -> row (v or v+8), col = lane&15
#pragma unroll
  for (int t = 0; t < NT; ++t) {
    int ncol = (tn0 + t) * 16 + l15 + colOff;
#pragma unroll
    for (int v = 0; v < 8; ++v) {
      int pr = tm * 16 + (half ? v + 8 : v);
      if (pr < npairs) {
        int o = oi[pr];
        float* pp = Out + (long long)o * ldOut + ncol;
        *pp = *pp + acc[t][v];
      }
    }
  }
}

// ---------------------------------------------------------------------------
// Utility kernels (grid-stride; incremental div/mod -> no per-iter division)
// ---------------------------------------------------------------------------
__global__ void zero_f32_kernel(float* __restrict__ p, long long n) {
  long long i = (long long)blockIdx.x * blockDim.x + threadIdx.x;
  long long s = (long long)gridDim.x * blockDim.x;
  for (; i < n; i += s) p[i] = 0.f;
}

// W (K, Cin, Cout) f32  ->  Wt (K, Cout, Cpad) f16 transposed + zero-padded
__global__ void cast_wt_kernel(const float* __restrict__ W, _Float16* __restrict__ Wt,
                               int Cin, int Cout, int Cpad, long long total) {
  long long i = (long long)blockIdx.x * blockDim.x + threadIdx.x;
  long long s = (long long)gridDim.x * blockDim.x;
  for (; i < total; i += s) {
    int c = (int)(i % Cpad);
    long long rem = i / Cpad;
    int n = (int)(rem % Cout);
    long long k = rem / Cout;
    float v = (c < Cin) ? W[(k * Cin + c) * Cout + n] : 0.f;
    Wt[i] = (_Float16)v;
  }
}

// Per-channel sum / sum-of-squares partials via LDS atomics -> global atomics.
__global__ __launch_bounds__(256)
void bn_partial_kernel(const float* __restrict__ X, long long total, int C,
                       float* __restrict__ sums, float* __restrict__ sumsq) {
  __shared__ float s1[256];
  __shared__ float s2[256];
  for (int c = threadIdx.x; c < C; c += blockDim.x) { s1[c] = 0.f; s2[c] = 0.f; }
  __syncthreads();
  long long i = (long long)blockIdx.x * blockDim.x + threadIdx.x;
  long long s = (long long)gridDim.x * blockDim.x;
  int c  = (int)(i % C);
  int cs = (int)(s % C);
  for (; i < total; i += s) {
    float v = X[i];
    atomicAdd(&s1[c], v);
    atomicAdd(&s2[c], v * v);
    c += cs;
    if (c >= C) c -= C;
  }
  __syncthreads();
  for (int c2 = threadIdx.x; c2 < C; c2 += blockDim.x) {
    atomicAdd(&sums[c2], s1[c2]);
    atomicAdd(&sumsq[c2], s2[c2]);
  }
}

__global__ void bn_finalize_kernel(const float* __restrict__ sums,
                                   const float* __restrict__ sumsq,
                                   const float* __restrict__ g,
                                   const float* __restrict__ b,
                                   float* __restrict__ scale,
                                   float* __restrict__ shift,
                                   int C, float invN) {
  int c = blockIdx.x * blockDim.x + threadIdx.x;
  if (c < C) {
    float mu  = sums[c] * invN;
    float var = sumsq[c] * invN - mu * mu;
    float sc  = g[c] * rsqrtf(var + 1e-4f);
    scale[c] = sc;
    shift[c] = b[c] - mu * sc;
  }
}

// fused BN(scale,shift) + ReLU + f32->f16 cast, writing K-padded (zeros) rows.
// scale==nullptr -> identity (plain cast), used for the raw input features.
__global__ void bnrelu_cast_kernel(const float* __restrict__ X, _Float16* __restrict__ Y,
                                   long long N, int C, int Cpad,
                                   const float* __restrict__ scale,
                                   const float* __restrict__ shift, int relu) {
  long long total = N * Cpad;
  long long i = (long long)blockIdx.x * blockDim.x + threadIdx.x;
  long long s = (long long)gridDim.x * blockDim.x;
  long long r  = i / Cpad;
  int       c  = (int)(i % Cpad);
  long long rs = s / Cpad;
  int       cs = (int)(s % Cpad);
  for (; i < total; i += s) {
    float v = 0.f;
    if (c < C) {
      v = X[r * C + c];
      if (scale) v = fmaf(v, scale[c], shift[c]);
      if (relu)  v = fmaxf(v, 0.f);
    }
    Y[i] = (_Float16)v;
    c += cs;
    r += rs;
    if (c >= Cpad) { c -= Cpad; r += 1; }
  }
}

// BN + ReLU staying in f32 (final head input).
__global__ void bnrelu_f32_kernel(const float* __restrict__ X, float* __restrict__ Y,
                                  long long N, int C,
                                  const float* __restrict__ scale,
                                  const float* __restrict__ shift) {
  long long total = N * C;
  long long i = (long long)blockIdx.x * blockDim.x + threadIdx.x;
  long long s = (long long)gridDim.x * blockDim.x;
  int c  = (int)(i % C);
  int cs = (int)(s % C);
  for (; i < total; i += s) {
    float v = fmaf(X[i], scale[c], shift[c]);
    Y[i] = fmaxf(v, 0.f);
    c += cs;
    if (c >= C) c -= C;
  }
}

// copy src (N x C) into dst (N x ldDst) at column colOff (skip-concat)
__global__ void copycols_kernel(const float* __restrict__ src, float* __restrict__ dst,
                                long long N, int C, int ldDst, int colOff) {
  long long total = N * C;
  long long i = (long long)blockIdx.x * blockDim.x + threadIdx.x;
  long long s = (long long)gridDim.x * blockDim.x;
  long long r  = i / C;
  int       c  = (int)(i % C);
  long long rs = s / C;
  int       cs = (int)(s % C);
  for (; i < total; i += s) {
    dst[r * ldDst + colOff + c] = src[i];
    c += cs;
    r += rs;
    if (c >= C) { c -= C; r += 1; }
  }
}

// final linear head: out (N x 20) = X (N x 16) @ Wl (16 x 20) + bl
__global__ void linear_kernel(const float* __restrict__ X, const float* __restrict__ Wl,
                              const float* __restrict__ bl, float* __restrict__ out,
                              long long N) {
  long long total = N * 20;
  long long i = (long long)blockIdx.x * blockDim.x + threadIdx.x;
  long long s = (long long)gridDim.x * blockDim.x;
  long long r  = i / 20;
  int       j  = (int)(i % 20);
  long long rs = s / 20;
  int       js = (int)(s % 20);
  for (; i < total; i += s) {
    float acc = bl[j];
#pragma unroll
    for (int c = 0; c < 16; ++c)
      acc = fmaf(X[r * 16 + c], Wl[c * 20 + j], acc);
    out[i] = acc;
    j += js;
    r += rs;
    if (j >= 20) { j -= 20; r += 1; }
  }
}

// ---------------------------------------------------------------------------
// Host orchestration
// ---------------------------------------------------------------------------
extern "C" void kernel_launch(void* const* d_in, const int* in_sizes, int n_in,
                              void* d_out, int out_size, void* d_ws, size_t ws_size,
                              hipStream_t stream) {
  (void)n_in; (void)out_size; (void)ws_size;
  constexpr int Lv = 7;

  // ---- decode flattened input pytree (setup_inputs dict insertion order) ----
  int p = 0;
  const float* feats = (const float*)d_in[p++];
  const float* Wstem = (const float*)d_in[p++];

  const float *bn0g[Lv], *bn0b[Lv], *W0[Lv];
  const float *bndg[Lv] = {}, *bndb[Lv] = {}, *Wd[Lv] = {};
  const float *bnug[Lv] = {}, *bnub[Lv] = {}, *Wu[Lv] = {};
  const float *bnpg[Lv] = {}, *bnpb[Lv] = {}, *Wp[Lv] = {};
  for (int l = 0; l < Lv; ++l) {
    bn0g[l] = (const float*)d_in[p++];
    bn0b[l] = (const float*)d_in[p++];
    W0[l]   = (const float*)d_in[p++];
    if (l < Lv - 1) {
      bndg[l] = (const float*)d_in[p++];
      bndb[l] = (const float*)d_in[p++];
      Wd[l]   = (const float*)d_in[p++];
      bnug[l] = (const float*)d_in[p++];
      bnub[l] = (const float*)d_in[p++];
      Wu[l]   = (const float*)d_in[p++];
      bnpg[l] = (const float*)d_in[p++];
      bnpb[l] = (const float*)d_in[p++];
      Wp[l]   = (const float*)d_in[p++];
    }
  }
  const float* bnfg = (const float*)d_in[p++];
  const float* bnfb = (const float*)d_in[p++];
  const float* Wlin = (const float*)d_in[p++];
  const float* blin = (const float*)d_in[p++];

  const int *smii[Lv][27], *smoi[Lv][27];
  int smn[Lv][27];
  for (int l = 0; l < Lv; ++l)
    for (int k = 0; k < 27; ++k) {
      smii[l][k] = (const int*)d_in[p];
      smn[l][k]  = in_sizes[p];
      p++;
      smoi[l][k] = (const int*)d_in[p++];
    }
  const int *dnfi[Lv - 1][8], *dnci[Lv - 1][8];
  int dnn[Lv - 1][8];
  for (int l = 0; l < Lv - 1; ++l)
    for (int k = 0; k < 8; ++k) {
      dnfi[l][k] = (const int*)d_in[p];
      dnn[l][k]  = in_sizes[p];
      p++;
      dnci[l][k] = (const int*)d_in[p++];
    }
  // counts (device scalars) skipped; recover on host from the center offset
  // of the submanifold rulebook (offset 13 = (0,0,0) pairs every voxel).
  int cnt[Lv];
  for (int l = 0; l < Lv; ++l) cnt[l] = smn[l][13];
  int aw[Lv];
  for (int l = 0; l < Lv; ++l) aw[l] = 16 * (l + 1);
  auto pad32 = [](int c) { return (c + 31) & ~31; };

  // ---- workspace bump allocator ----
  char* base = (char*)d_ws;
  size_t off = 0;
  auto alloc = [&](size_t bytes) -> void* {
    off = (off + 255) & ~(size_t)255;
    void* r = base + off;
    off += bytes;
    return r;
  };

  // f16 transposed/padded weights
  _Float16* stemWt = (_Float16*)alloc((size_t)27 * 16 * 32 * sizeof(_Float16));
  _Float16 *W0t[Lv], *Wdt[Lv - 1], *Wut[Lv - 1], *Wpt[Lv - 1];
  for (int l = 0; l < Lv; ++l)
    W0t[l] = (_Float16*)alloc((size_t)27 * aw[l] * pad32(aw[l]) * sizeof(_Float16));
  for (int l = 0; l < Lv - 1; ++l) {
    int a = aw[l], b = aw[l + 1];
    Wdt[l] = (_Float16*)alloc((size_t)8 * b * pad32(a) * sizeof(_Float16));
    Wut[l] = (_Float16*)alloc((size_t)8 * a * pad32(b) * sizeof(_Float16));
    Wpt[l] = (_Float16*)alloc((size_t)27 * a * pad32(2 * a) * sizeof(_Float16));
  }

  // f32 activations
  float *act[Lv], *xcv[Lv], *catb[Lv - 1], *res[Lv];
  for (int l = 0; l < Lv; ++l) {
    act[l] = (float*)alloc((size_t)cnt[l] * aw[l] * sizeof(float));
    xcv[l] = (float*)alloc((size_t)cnt[l] * aw[l] * sizeof(float));
  }
  for (int l = 0; l < Lv - 1; ++l) {
    catb[l] = (float*)alloc((size_t)cnt[l] * 2 * aw[l] * sizeof(float));
    res[l]  = (float*)alloc((size_t)cnt[l] * aw[l] * sizeof(float));
  }
  res[Lv - 1] = xcv[Lv - 1];
  float* finalbuf = (float*)alloc((size_t)cnt[0] * 16 * sizeof(float));

  float* sums  = (float*)alloc(512 * sizeof(float));  // sums[256] + sumsq[256]
  float* sumsq = sums + 256;
  float* scale = (float*)alloc(256 * sizeof(float));
  float* shift = (float*)alloc(256 * sizeof(float));

  // shared f16 activation scratch sized for the worst user
  size_t f16elems = (size_t)cnt[0] * 32;  // feats pad32(3)=32
  for (int l = 0; l < Lv; ++l) {
    int cmax = (l < Lv - 1) ? 2 * aw[l] : aw[l];
    size_t e = (size_t)cnt[l] * pad32(cmax);
    if (e > f16elems) f16elems = e;
  }
  _Float16* xf16 = (_Float16*)alloc(f16elems * sizeof(_Float16));

  // ---- launch helpers ----
  auto gsb = [](long long tot) {
    long long b = (tot + 255) / 256;
    if (b < 1) b = 1;
    if (b > 16384) b = 16384;
    return (int)b;
  };
  auto zero = [&](float* ptr, long long n) {
    zero_f32_kernel<<<gsb(n), 256, 0, stream>>>(ptr, n);
  };
  auto castW = [&](const float* W, _Float16* Wt, int K, int Cin, int Cout) {
    long long total = (long long)K * Cout * ((Cin + 31) & ~31);
    cast_wt_kernel<<<gsb(total), 256, 0, stream>>>(W, Wt, Cin, Cout,
                                                   (Cin + 31) & ~31, total);
  };
  auto bnstats = [&](const float* X, int N, int C, const float* g, const float* b) {
    zero(sums, 512);
    long long total = (long long)N * C;
    int nb = gsb(total);
    if (nb > 2048) nb = 2048;
    bn_partial_kernel<<<nb, 256, 0, stream>>>(X, total, C, sums, sumsq);
    bn_finalize_kernel<<<1, 256, 0, stream>>>(sums, sumsq, g, b, scale, shift, C,
                                              1.f / (float)N);
  };
  auto applyCast = [&](const float* X, int N, int C, const float* sc,
                       const float* sh, int relu) {
    long long total = (long long)N * pad32(C);
    bnrelu_cast_kernel<<<gsb(total), 256, 0, stream>>>(X, xf16, N, C, pad32(C),
                                                       sc, sh, relu);
  };
  auto conv = [&](int ldX, const _Float16* Wt, const int* ii, const int* oi,
                  int n, int Cout, float* Out, int ldOut, int colOff) {
    if (n <= 0) return;
    int nt16 = Cout / 16;
    if ((nt16 % 4) == 0) {
      dim3 grid((n + 15) / 16, nt16 / 4);
      gsconv_wmma_kernel<4><<<grid, 32, 0, stream>>>(xf16, ldX, Wt, ii, oi, n,
                                                     Out, ldOut, colOff);
    } else if ((nt16 % 2) == 0) {
      dim3 grid((n + 15) / 16, nt16 / 2);
      gsconv_wmma_kernel<2><<<grid, 32, 0, stream>>>(xf16, ldX, Wt, ii, oi, n,
                                                     Out, ldOut, colOff);
    } else {
      dim3 grid((n + 15) / 16, nt16);
      gsconv_wmma_kernel<1><<<grid, 32, 0, stream>>>(xf16, ldX, Wt, ii, oi, n,
                                                     Out, ldOut, colOff);
    }
  };

  // ---- 0) weight precasting (every call; deterministic, no caching) ----
  castW(Wstem, stemWt, 27, 3, 16);
  for (int l = 0; l < Lv; ++l) castW(W0[l], W0t[l], 27, aw[l], aw[l]);
  for (int l = 0; l < Lv - 1; ++l) {
    castW(Wd[l], Wdt[l], 8, aw[l], aw[l + 1]);
    castW(Wu[l], Wut[l], 8, aw[l + 1], aw[l]);
    castW(Wp[l], Wpt[l], 27, 2 * aw[l], aw[l]);
  }

  // ---- 1) stem: feats (N0 x 3) -> act[0] (N0 x 16) ----
  applyCast(feats, cnt[0], 3, nullptr, nullptr, 0);
  zero(act[0], (long long)cnt[0] * 16);
  for (int k = 0; k < 27; ++k)
    conv(32, stemWt + (size_t)k * 16 * 32, smii[0][k], smoi[0][k], smn[0][k], 16,
         act[0], 16, 0);

  // ---- 2) down path ----
  for (int l = 0; l < Lv; ++l) {
    int a = aw[l], N = cnt[l], pa = pad32(a);
    bnstats(act[l], N, a, bn0g[l], bn0b[l]);
    applyCast(act[l], N, a, scale, shift, 1);
    zero(xcv[l], (long long)N * a);
    for (int k = 0; k < 27; ++k)
      conv(pa, W0t[l] + (size_t)k * a * pa, smii[l][k], smoi[l][k], smn[l][k], a,
           xcv[l], a, 0);
    if (l < Lv - 1) {
      int b = aw[l + 1];
      bnstats(xcv[l], N, a, bndg[l], bndb[l]);
      applyCast(xcv[l], N, a, scale, shift, 1);
      zero(act[l + 1], (long long)cnt[l + 1] * b);
      for (int k = 0; k < 8; ++k)
        conv(pa, Wdt[l] + (size_t)k * b * pa, dnfi[l][k], dnci[l][k], dnn[l][k], b,
             act[l + 1], b, 0);
    }
  }

  // ---- 3) up path ----
  for (int l = Lv - 2; l >= 0; --l) {
    int a = aw[l], b = aw[l + 1], N = cnt[l], Nc = cnt[l + 1];
    int pb = pad32(b), p2a = pad32(2 * a);
    const float* y = (l == Lv - 2) ? xcv[Lv - 1] : res[l + 1];
    bnstats(y, Nc, b, bnug[l], bnub[l]);
    applyCast(y, Nc, b, scale, shift, 1);
    zero(catb[l], (long long)N * 2 * a);
    copycols_kernel<<<gsb((long long)N * a), 256, 0, stream>>>(xcv[l], catb[l], N,
                                                               a, 2 * a, 0);
    for (int k = 0; k < 8; ++k)  // deconv: gather coarse (ci), scatter fine (fi)
      conv(pb, Wut[l] + (size_t)k * a * pb, dnci[l][k], dnfi[l][k], dnn[l][k], a,
           catb[l], 2 * a, a);
    bnstats(catb[l], N, 2 * a, bnpg[l], bnpb[l]);
    applyCast(catb[l], N, 2 * a, scale, shift, 1);
    zero(res[l], (long long)N * a);
    for (int k = 0; k < 27; ++k)
      conv(p2a, Wpt[l] + (size_t)k * a * p2a, smii[l][k], smoi[l][k], smn[l][k], a,
           res[l], a, 0);
  }

  // ---- 4) final BN+ReLU + linear head ----
  bnstats(res[0], cnt[0], 16, bnfg, bnfb);
  bnrelu_f32_kernel<<<gsb((long long)cnt[0] * 16), 256, 0, stream>>>(
      res[0], finalbuf, cnt[0], 16, scale, shift);
  linear_kernel<<<gsb((long long)cnt[0] * 20), 256, 0, stream>>>(
      finalbuf, Wlin, blin, (float*)d_out, cnt[0]);
}
